// GINWrapper_53257594471011
// MI455X (gfx1250) — matrix operand order repaired and checked by reading, not compiled, not run on hardware
//
#include <hip/hip_runtime.h>
#include <stdint.h>

#define N_NODES  100000
#define N_EDGES  625000
#define D        128      // D_IN == D_HID
#define EDGE_DIM 64

typedef __attribute__((ext_vector_type(16))) __bf16 v16bf;
typedef __attribute__((ext_vector_type(8)))  float  v8f;

__device__ __forceinline__ float relu_f(float v) { return v > 0.0f ? v : 0.0f; }

// ---------------------------------------------------------------------------
// Zero the aggregation workspace (must run every launch; ws is not re-poisoned)
// ---------------------------------------------------------------------------
__global__ void zero_ws_kernel(float4* __restrict__ p, int n4) {
    int i      = blockIdx.x * blockDim.x + threadIdx.x;
    int stride = gridDim.x * blockDim.x;
    for (; i < n4; i += stride) p[i] = make_float4(0.f, 0.f, 0.f, 0.f);
}

// ---------------------------------------------------------------------------
// Edge pass: e = edge_attr @ We + be ; msg = relu(x[src] + e) ; agg[dst] += msg
// One wave handles a 16-edge x 128-dim tile with v_wmma_f32_16x16x32_bf16.
// Gathers/scatters use SGPR-base + u32-VGPR-offset (scale_offset) addressing;
// epilogue gathers are batched 8-wide into distinct temps for memory-level
// parallelism (avoids one-register load->wait->use serialization).
// ---------------------------------------------------------------------------
__global__ void __launch_bounds__(128) gine_edge_kernel(
    const float* __restrict__ x,
    const int*   __restrict__ edge_index,
    const float* __restrict__ edge_attr,
    const float* __restrict__ We,
    const float* __restrict__ be,
    float*       __restrict__ agg)
{
    __shared__ v16bf ldsB[8][2][32];   // [n_tile][k_step][lane], 16 KB

    const int  tid  = threadIdx.x;
    const int  lane = tid & 31;
    const int  wave = tid >> 5;
    const bool hi   = (lane >= 16);
    const int  nlo  = lane & 15;

    // ---- stage We (64x128 f32) -> bf16 B fragments in LDS ----
    for (int idx = tid; idx < 8 * 2 * 32; idx += blockDim.x) {
        int l  = idx & 31;
        int s  = (idx >> 5) & 1;
        int t  = idx >> 6;
        int n  = t * 16 + (l & 15);
        int kb = s * 32 + ((l >= 16) ? 16 : 0);
        v16bf frag;
        #pragma unroll
        for (int j = 0; j < 16; ++j)
            frag[j] = (__bf16)We[(kb + j) * D + n];
        ldsB[t][s][l] = frag;
    }
    __syncthreads();

    const int  nFull      = N_EDGES / 16;             // 39062 full tiles
    const int  waveGlobal = blockIdx.x * 4 + wave;
    const int  nWaves     = gridDim.x * 4;
    const int* srcIdx     = edge_index;
    const int* dstIdx     = edge_index + N_EDGES;

    // tile-invariant bias values for this lane's column (d = t*16 + nlo)
    float bias[8];
    #pragma unroll
    for (int t = 0; t < 8; ++t) bias[t] = be[t * 16 + nlo];

    for (int tile = waveGlobal; tile < nFull; tile += nWaves) {
        const int eBase = tile * 16;

        // ---- A fragments: 16 edges x 64 (row M = lane&15), u32 offset ----
        const uint32_t abase = (uint32_t)(eBase + nlo) * EDGE_DIM;
        v16bf afrag[2];
        #pragma unroll
        for (int s = 0; s < 2; ++s) {
            const int kb = s * 32 + (hi ? 16 : 0);
            #pragma unroll
            for (int j = 0; j < 16; ++j) {
                const int k = kb + j;
                afrag[s][j] = (__bf16)edge_attr[(size_t)abase + k];
            }
        }

        // ---- per-row (M = m + hi*8) 32-bit gather/scatter element offsets ----
        uint32_t xoff[8];
        uint32_t aoff[8];
        #pragma unroll
        for (int m = 0; m < 8; ++m) {
            const int e = eBase + m + (hi ? 8 : 0);
            xoff[m] = (uint32_t)srcIdx[e] * D + nlo;
            aoff[m] = (uint32_t)dstIdx[e] * D + nlo;
        }

        // ---- 8 N-tiles: WMMA accumulate then fused epilogue ----
        #pragma unroll
        for (int t = 0; t < 8; ++t) {
            v8f acc = {};
            acc = __builtin_amdgcn_wmma_f32_16x16x32_bf16(
                      false, afrag[0], false, ldsB[t][0][lane], (short)0, acc, false, false);
            acc = __builtin_amdgcn_wmma_f32_16x16x32_bf16(
                      false, afrag[1], false, ldsB[t][1][lane], (short)0, acc, false, false);

            // batch the 8 gathers into distinct temps -> 8 loads in flight
            float g[8];
            #pragma unroll
            for (int m = 0; m < 8; ++m)
                g[m] = x[(size_t)xoff[m] + t * 16];

            #pragma unroll
            for (int m = 0; m < 8; ++m) {
                float v = relu_f(acc[m] + bias[t] + g[m]);
                atomicAdd(&agg[(size_t)aoff[m] + t * 16], v);
            }
        }
    }

    // ---- scalar fp32 tail: edges [nFull*16, N_EDGES), handled by one wave ----
    if (waveGlobal == 0) {
        const int tailBase = nFull * 16;               // 624992, 8 edges left
        const int e        = tailBase + (lane >> 2);   // 4 lanes per edge
        if (e < N_EDGES) {
            const int    s  = srcIdx[e];
            const int    dt = dstIdx[e];
            const float* er = edge_attr + (size_t)e * EDGE_DIM;
            const int    d0 = (lane & 3) * 32;         // 32 dims per lane
            for (int d = d0; d < d0 + 32; ++d) {
                float acc = be[d];
                #pragma unroll
                for (int k = 0; k < EDGE_DIM; ++k)
                    acc += er[k] * We[k * D + d];
                float v = relu_f(acc + x[(size_t)s * D + d]);
                atomicAdd(&agg[(size_t)dt * D + d], v);
            }
        }
    }
}

// ---------------------------------------------------------------------------
// Node pass: out = relu( (x + agg) @ Wm + bm )   (wrapper relu is idempotent)
// One wave handles a 16-node x 128-dim tile; Wm staged in LDS as B fragments.
// ---------------------------------------------------------------------------
__global__ void __launch_bounds__(128) gine_node_kernel(
    const float* __restrict__ x,
    const float* __restrict__ agg,
    const float* __restrict__ Wm,
    const float* __restrict__ bm,
    float*       __restrict__ out)
{
    __shared__ v16bf ldsB[8][4][32];   // [n_tile][k_step][lane], 32 KB

    const int  tid  = threadIdx.x;
    const int  lane = tid & 31;
    const int  wave = tid >> 5;
    const bool hi   = (lane >= 16);
    const int  nlo  = lane & 15;

    // ---- stage Wm (128x128 f32) -> bf16 B fragments in LDS ----
    for (int idx = tid; idx < 8 * 4 * 32; idx += blockDim.x) {
        int l  = idx & 31;
        int s  = (idx >> 5) & 3;
        int t  = idx >> 7;
        int n  = t * 16 + (l & 15);
        int kb = s * 32 + ((l >= 16) ? 16 : 0);
        v16bf frag;
        #pragma unroll
        for (int j = 0; j < 16; ++j)
            frag[j] = (__bf16)Wm[(kb + j) * D + n];
        ldsB[t][s][l] = frag;
    }
    __syncthreads();

    const int nTiles     = N_NODES / 16;  // 6250 exactly, no tail
    const int waveGlobal = blockIdx.x * 4 + wave;
    const int nWaves     = gridDim.x * 4;

    float bias[8];
    #pragma unroll
    for (int t = 0; t < 8; ++t) bias[t] = bm[t * 16 + nlo];

    for (int tile = waveGlobal; tile < nTiles; tile += nWaves) {
        // u32 element offset of this lane's A-row
        const uint32_t rbase = (uint32_t)(tile * 16 + nlo) * D;

        // ---- A fragments: h = x + agg, converted to bf16 ----
        v16bf afrag[4];
        #pragma unroll
        for (int s = 0; s < 4; ++s) {
            const int kb = s * 32 + (hi ? 16 : 0);
            #pragma unroll
            for (int j = 0; j < 16; ++j) {
                const int k = kb + j;
                afrag[s][j] = (__bf16)(x[(size_t)rbase + k] + agg[(size_t)rbase + k]);
            }
        }

        // ---- per-row output u32 element offsets (M = m + hi*8) ----
        uint32_t ooff[8];
        #pragma unroll
        for (int m = 0; m < 8; ++m)
            ooff[m] = (uint32_t)(tile * 16 + m + (hi ? 8 : 0)) * D + nlo;

        // ---- 8 N-tiles x 4 K-steps of WMMA, relu + store ----
        #pragma unroll
        for (int t = 0; t < 8; ++t) {
            v8f acc = {};
            #pragma unroll
            for (int s = 0; s < 4; ++s)
                acc = __builtin_amdgcn_wmma_f32_16x16x32_bf16(
                          false, afrag[s], false, ldsB[t][s][lane], (short)0, acc, false, false);
            #pragma unroll
            for (int m = 0; m < 8; ++m)
                out[(size_t)ooff[m] + t * 16] = relu_f(acc[m] + bias[t]);
        }
    }
}

// ---------------------------------------------------------------------------
extern "C" void kernel_launch(void* const* d_in, const int* in_sizes, int n_in,
                              void* d_out, int out_size, void* d_ws, size_t ws_size,
                              hipStream_t stream) {
    const float* x   = (const float*)d_in[0];   // [100000,128] f32
    const int*   ei  = (const int*)  d_in[1];   // [2,625000]  int32
    const float* ea  = (const float*)d_in[2];   // [625000,64] f32
    const float* We  = (const float*)d_in[3];   // [64,128]    f32
    const float* be  = (const float*)d_in[4];   // [128]       f32
    const float* Wm  = (const float*)d_in[5];   // [128,128]   f32
    const float* bm  = (const float*)d_in[6];   // [128]       f32
    float*       out = (float*)d_out;           // [100000,128] f32
    float*       agg = (float*)d_ws;            // 51.2 MB scratch accumulator

    const int n4 = (N_NODES * D) / 4;
    zero_ws_kernel<<<1024, 256, 0, stream>>>((float4*)agg, n4);
    gine_edge_kernel<<<1280, 128, 0, stream>>>(x, ei, ea, We, be, agg);
    gine_node_kernel<<<640, 128, 0, stream>>>(x, agg, Wm, bm, out);
}
